// BiMambaBlock_48473000902760
// MI455X (gfx1250) — matrix operand order repaired and checked by reading, not compiled, not run on hardware
//
#include <hip/hip_runtime.h>
#include <cstdint>

// B=32, S=2048, D=256. Bidirectional GRU block.
// Pipeline: convert weights f32->f16, LayerNorm, WMMA input projections,
// persistent WMMA recurrent scan (4 WGs; h written out via
// global_store_async_from_lds_b128), fused gate+output WMMA kernel.

#define Bb 32
#define Ss 2048
#define Dd 256
#define NROWS (Bb * Ss)          // 65536
#define H3 (3 * Dd)              // 768

typedef __attribute__((ext_vector_type(16))) _Float16 v16h;
typedef __attribute__((ext_vector_type(8)))  _Float16 v8h;
typedef __attribute__((ext_vector_type(8)))  float    v8f;

__device__ __forceinline__ v8f wmma16(v16h a, v16h b, v8f c) {
  return __builtin_amdgcn_wmma_f32_16x16x32_f16(false, a, false, b, (short)0, c,
                                                false, false);
}

__device__ __forceinline__ v8f zero8() {
  v8f z = {0.f, 0.f, 0.f, 0.f, 0.f, 0.f, 0.f, 0.f};
  return z;
}

__device__ __forceinline__ float sigf(float x) {
  return 1.f / (1.f + __expf(-x));
}
__device__ __forceinline__ float tanh_fast(float x) {
  x = fminf(fmaxf(x, -15.f), 15.f);
  float e = __expf(-2.f * x);
  return (1.f - e) / (1.f + e);
}

// A-fragment (16x32 f16) from row-major [M][K] buffer (ld in halves).
// ISA layout: lane<16 holds K = k0+0..7 and k0+16..23; lane>=16 holds
// K = k0+8..15 and k0+24..31; M = lane&15.
__device__ __forceinline__ v16h load_a_frag(const _Float16* base, int ld,
                                            int k0, int lane) {
  int m  = lane & 15;
  int ks = (lane >> 4) << 3;  // 0 or 8
  const _Float16* p = base + m * ld + k0 + ks;
  v8h lo = *(const v8h*)(p);
  v8h hi = *(const v8h*)(p + 16);
  return __builtin_shufflevector(lo, hi, 0, 1, 2, 3, 4, 5, 6, 7,
                                 8, 9, 10, 11, 12, 13, 14, 15);
}

// B-fragment (32x16 f16) from row-major [N][K] buffer (ld in halves).
// lane holds column N = n0+(lane&15), K = k0 + (lane<16 ? 0..7,16..23 : 8..15,24..31).
__device__ __forceinline__ v16h load_b_frag(const _Float16* w, int ld,
                                            int n0, int k0, int lane) {
  int n  = lane & 15;
  int kb = (lane >> 4) << 3;  // 0 or 8
  const _Float16* p = w + (size_t)(n0 + n) * ld + k0 + kb;
  v8h lo = *(const v8h*)(p);
  v8h hi = *(const v8h*)(p + 16);
  return __builtin_shufflevector(lo, hi, 0, 1, 2, 3, 4, 5, 6, 7,
                                 8, 9, 10, 11, 12, 13, 14, 15);
}

// ---------------------------------------------------------------- converts
__global__ void f32_to_f16_kernel(const float* __restrict__ src,
                                  _Float16* __restrict__ dst, int n) {
  int i = blockIdx.x * blockDim.x + threadIdx.x;
  if (i < n) dst[i] = (_Float16)src[i];
}

// ---------------------------------------------------------------- layernorm
// One wave per row of 256; 8 elements per lane; shfl_xor reductions (wave32).
__global__ __launch_bounds__(256) void ln_kernel(const float* __restrict__ x,
                                                 const float* __restrict__ g,
                                                 const float* __restrict__ b,
                                                 _Float16* __restrict__ xn) {
  int wave = threadIdx.x >> 5, lane = threadIdx.x & 31;
  int row = blockIdx.x * 8 + wave;
  const float* xr = x + (size_t)row * Dd + lane * 8;
  float v[8];
#pragma unroll
  for (int k = 0; k < 8; ++k) v[k] = xr[k];
  float s = 0.f;
#pragma unroll
  for (int k = 0; k < 8; ++k) s += v[k];
#pragma unroll
  for (int o = 16; o > 0; o >>= 1) s += __shfl_xor(s, o, 32);
  float mu = s * (1.f / 256.f);
  float vs = 0.f;
#pragma unroll
  for (int k = 0; k < 8; ++k) { float d = v[k] - mu; vs += d * d; }
#pragma unroll
  for (int o = 16; o > 0; o >>= 1) vs += __shfl_xor(vs, o, 32);
  float inv = rsqrtf(vs * (1.f / 256.f) + 1e-5f);
  v8h out;
#pragma unroll
  for (int k = 0; k < 8; ++k) {
    int c = lane * 8 + k;
    out[k] = (_Float16)((v[k] - mu) * inv * g[c] + b[c]);
  }
  *(v8h*)(xn + (size_t)row * Dd + lane * 8) = out;
}

// ---------------------------------------------------------------- xp GEMM
// xp[d] = xn @ w_ih[d]^T + b_ih[d]; M=65536, N=768, K=256. 16 waves/block,
// wave w computes N-tiles {w, w+16, w+32} for the block's 16-row A tile.
__global__ __launch_bounds__(512) void proj_kernel(
    const _Float16* __restrict__ xn,
    const _Float16* __restrict__ wih_f, const _Float16* __restrict__ wih_r,
    const float* __restrict__ bih_f, const float* __restrict__ bih_r,
    _Float16* __restrict__ xp_f, _Float16* __restrict__ xp_r) {
  __shared__ __attribute__((aligned(32))) _Float16 aLds[16 * Dd];
  int dir = blockIdx.y;
  const _Float16* w  = dir ? wih_r : wih_f;
  const float*    bi = dir ? bih_r : bih_f;
  _Float16*       xp = dir ? xp_r : xp_f;

  int row0 = blockIdx.x * 16;
  int tid = threadIdx.x;
  // stage A: 16 rows x 256 halves = 512 v8h chunks
  ((v8h*)aLds)[tid] = ((const v8h*)(xn + (size_t)row0 * Dd))[tid];
  __syncthreads();

  int wv = tid >> 5, lane = tid & 31, nl = lane & 15;
  int mhi = (lane >> 4) << 3;
  v16h a[8];
#pragma unroll
  for (int ks = 0; ks < 8; ++ks) a[ks] = load_a_frag(aLds, Dd, ks * 32, lane);

#pragma unroll
  for (int p = 0; p < 3; ++p) {
    int n0 = p * 256 + wv * 16;
    v8f acc = zero8();
#pragma unroll
    for (int ks = 0; ks < 8; ++ks)
      acc = wmma16(a[ks], load_b_frag(w, Dd, n0, ks * 32, lane), acc);
    int j = n0 + nl;
    float bias = bi[j];
#pragma unroll
    for (int r = 0; r < 8; ++r) {
      int m = r + mhi;
      xp[(size_t)(row0 + m) * H3 + j] = (_Float16)(acc[r] + bias);
    }
  }
}

// ---------------------------------------------------------------- GRU scan
// 4 persistent WGs: (dir, batch-tile). 16 waves; wave w owns hidden tile w:
// 3 gate GEMM tiles/step; h lives in double-buffered LDS and is drained to
// global via async LDS->global DMA (ASYNCcnt), off the serial critical path.
__global__ __launch_bounds__(512, 1) void scan_kernel(
    const _Float16* __restrict__ xp_f, const _Float16* __restrict__ xp_r,
    const _Float16* __restrict__ whh_f, const _Float16* __restrict__ whh_r,
    const float* __restrict__ bhh_f, const float* __restrict__ bhh_r,
    _Float16* __restrict__ h_f, _Float16* __restrict__ h_r) {
  __shared__ __attribute__((aligned(32))) _Float16 hbuf[2][16 * Dd];

  int dir = blockIdx.x & 1, mt = blockIdx.x >> 1;
  const _Float16* xp  = dir ? xp_r : xp_f;
  const _Float16* whh = dir ? whh_r : whh_f;
  const float*    bhh = dir ? bhh_r : bhh_f;
  _Float16*       ho  = dir ? h_r : h_f;

  int tid = threadIdx.x, wv = tid >> 5, lane = tid & 31, nl = lane & 15;
  int mhi = (lane >> 4) << 3;
  int j = wv * 16 + nl;           // this lane's hidden column
  int b0 = mt * 16;

  // VGPR-resident recurrent weights: 3 gates x 8 k-steps (192 VGPRs)
  v16h bw[3][8];
#pragma unroll
  for (int p = 0; p < 3; ++p)
#pragma unroll
    for (int ks = 0; ks < 8; ++ks)
      bw[p][ks] = load_b_frag(whh, Dd, p * 256 + wv * 16, ks * 32, lane);
  float bh0 = bhh[j], bh1 = bhh[256 + j], bh2 = bhh[512 + j];

  // per-thread async-drain addressing: thread owns 16B chunk (sm, sc)
  int sm = tid >> 5;              // tile row 0..15
  int sc = (tid & 31) * 8;        // column start 0..248
  _Float16* ho_row = ho + (size_t)(b0 + sm) * Ss * Dd + sc;
  unsigned lds_src0 = (unsigned)(uintptr_t)&hbuf[0][sm * Dd + sc];
  unsigned lds_src1 = (unsigned)(uintptr_t)&hbuf[1][sm * Dd + sc];

  for (int i = tid; i < 16 * Dd; i += 512) hbuf[0][i] = (_Float16)0.f;
  v8f hprev = zero8();
  __syncthreads();

  for (int t = 0; t < Ss; ++t) {
    int cur = t & 1, nxt = cur ^ 1;
    int s_in = dir ? (Ss - 1 - t) : t;

    v16h a[8];
#pragma unroll
    for (int ks = 0; ks < 8; ++ks)
      a[ks] = load_a_frag(hbuf[cur], Dd, ks * 32, lane);

    v8f acc0 = zero8(), acc1 = zero8(), acc2 = zero8();
#pragma unroll
    for (int ks = 0; ks < 8; ++ks) {
      acc0 = wmma16(a[ks], bw[0][ks], acc0);
      acc1 = wmma16(a[ks], bw[1][ks], acc1);
      acc2 = wmma16(a[ks], bw[2][ks], acc2);
    }

    if (t + 1 < Ss) {  // prefetch next step's xp (global_prefetch_b8)
      int s2 = dir ? (Ss - 2 - t) : (t + 1);
      __builtin_prefetch(xp + ((size_t)(b0 + mhi) * Ss + s2) * H3 + j, 0, 1);
    }

#pragma unroll
    for (int r = 0; r < 8; ++r) {
      int m = r + mhi;
      size_t base = ((size_t)(b0 + m) * Ss + s_in) * H3 + j;
      float xr_ = (float)xp[base];
      float xz_ = (float)xp[base + 256];
      float xn_ = (float)xp[base + 512];
      float rg = sigf(acc0[r] + bh0 + xr_);
      float zg = sigf(acc1[r] + bh1 + xz_);
      float ng = tanh_fast(xn_ + rg * (acc2[r] + bh2));
      float hn = (1.f - zg) * ng + zg * hprev[r];
      hprev[r] = hn;
      hbuf[nxt][m * Dd + j] = (_Float16)hn;
    }

    // Drain-slot reuse safety: the async store issued two steps ago on slot
    // `nxt` (possibly by another wave) must be complete before this step's
    // writes above become visible for reuse. Each wave retires its own
    // ASYNCcnt here; the following barrier orders that wave-locally-complete
    // state across all waves before the next overwrite.
    asm volatile("s_wait_asynccnt 0x0" ::: "memory");
    __syncthreads();

    // hbuf[nxt] is complete: DMA the 8KB h-tile to global (coalesced 16B per
    // thread), tracked by ASYNCcnt, overlapping the next step's GEMM.
    {
      unsigned lsrc = nxt ? lds_src1 : lds_src0;
      uint64_t ga = (uint64_t)(uintptr_t)(ho_row + (size_t)s_in * Dd);
      asm volatile("global_store_async_from_lds_b128 %0, %1, off"
                   :: "v"(ga), "v"(lsrc) : "memory");
    }
  }
  // S_ENDPGM performs an implicit wait-idle, covering the final async stores.
}

// ------------------------------------------------- fused gate + out GEMMs
// g = sigmoid([hf|hr] @ gate_w^T + gate_b); comb = g*hf+(1-g)*hr (LDS-staged);
// out = comb @ out_w^T + out_b + x.
__global__ __launch_bounds__(512) void gate_out_kernel(
    const _Float16* __restrict__ h_f, const _Float16* __restrict__ h_r,
    const _Float16* __restrict__ gw, const float* __restrict__ gb,
    const _Float16* __restrict__ ow, const float* __restrict__ ob,
    const float* __restrict__ x, float* __restrict__ out) {
  __shared__ __attribute__((aligned(32))) _Float16 cat[16 * 512];
  __shared__ __attribute__((aligned(32))) _Float16 comb[16 * Dd];

  int row0 = blockIdx.x * 16;
  int tid = threadIdx.x;
  // stage [hf|hr]: 16 rows x 512 halves = 1024 v8h chunks
  for (int i = tid; i < 1024; i += 512) {
    int m = i >> 6, c8 = i & 63;
    const _Float16* src = (c8 < 32)
        ? (h_f + (size_t)(row0 + m) * Dd + c8 * 8)
        : (h_r + (size_t)(row0 + m) * Dd + (c8 - 32) * 8);
    ((v8h*)cat)[i] = *(const v8h*)src;
  }
  __syncthreads();

  int wv = tid >> 5, lane = tid & 31, nl = lane & 15;
  int mhi = (lane >> 4) << 3;
  int n0 = wv * 16, j = n0 + nl;

  // GEMM1: K=512
  v8f acc = zero8();
#pragma unroll
  for (int ks = 0; ks < 16; ++ks)
    acc = wmma16(load_a_frag(cat, 512, ks * 32, lane),
                 load_b_frag(gw, 512, n0, ks * 32, lane), acc);
  float gbias = gb[j];
#pragma unroll
  for (int r = 0; r < 8; ++r) {
    int m = r + mhi;
    float g = sigf(acc[r] + gbias);
    float hf = (float)cat[m * 512 + j];
    float hr = (float)cat[m * 512 + 256 + j];
    comb[m * Dd + j] = (_Float16)(g * hf + (1.f - g) * hr);
  }
  __syncthreads();

  // GEMM2: K=256
  v8f acc2 = zero8();
#pragma unroll
  for (int ks = 0; ks < 8; ++ks)
    acc2 = wmma16(load_a_frag(comb, Dd, ks * 32, lane),
                  load_b_frag(ow, Dd, n0, ks * 32, lane), acc2);
  float obias = ob[j];
#pragma unroll
  for (int r = 0; r < 8; ++r) {
    int m = r + mhi;
    size_t o = (size_t)(row0 + m) * Dd + j;
    out[o] = acc2[r] + obias + x[o];
  }
}

// ---------------------------------------------------------------- launcher
extern "C" void kernel_launch(void* const* d_in, const int* in_sizes, int n_in,
                              void* d_out, int out_size, void* d_ws,
                              size_t ws_size, hipStream_t stream) {
  (void)in_sizes; (void)n_in; (void)out_size; (void)ws_size;
  const float* x      = (const float*)d_in[0];
  const float* ln_g   = (const float*)d_in[1];
  const float* ln_b   = (const float*)d_in[2];
  const float* fwih   = (const float*)d_in[3];
  const float* fwhh   = (const float*)d_in[4];
  const float* fbih   = (const float*)d_in[5];
  const float* fbhh   = (const float*)d_in[6];
  const float* rwih   = (const float*)d_in[7];
  const float* rwhh   = (const float*)d_in[8];
  const float* rbih   = (const float*)d_in[9];
  const float* rbhh   = (const float*)d_in[10];
  const float* gate_w = (const float*)d_in[11];
  const float* gate_b = (const float*)d_in[12];
  const float* out_w  = (const float*)d_in[13];
  const float* out_b  = (const float*)d_in[14];
  float* out = (float*)d_out;

  // workspace carve-out (~304 MB), all chunks 256B-aligned
  char* w = (char*)d_ws;
  size_t off = 0;
  auto carve = [&](size_t bytes) -> char* {
    char* p = w + off;
    off += (bytes + 255) & ~(size_t)255;
    return p;
  };
  _Float16* xn16  = (_Float16*)carve((size_t)NROWS * Dd * 2);
  _Float16* xp_f  = (_Float16*)carve((size_t)NROWS * H3 * 2);
  _Float16* xp_r  = (_Float16*)carve((size_t)NROWS * H3 * 2);
  _Float16* h_f   = (_Float16*)carve((size_t)NROWS * Dd * 2);
  _Float16* h_r   = (_Float16*)carve((size_t)NROWS * Dd * 2);
  _Float16* wihf  = (_Float16*)carve((size_t)H3 * Dd * 2);
  _Float16* whhf  = (_Float16*)carve((size_t)H3 * Dd * 2);
  _Float16* wihr  = (_Float16*)carve((size_t)H3 * Dd * 2);
  _Float16* whhr  = (_Float16*)carve((size_t)H3 * Dd * 2);
  _Float16* gw16  = (_Float16*)carve((size_t)Dd * 2 * Dd * 2);
  _Float16* ow16  = (_Float16*)carve((size_t)Dd * Dd * 2);

  auto cvt = [&](const float* s, _Float16* d, int n) {
    f32_to_f16_kernel<<<(n + 255) / 256, 256, 0, stream>>>(s, d, n);
  };
  cvt(fwih, wihf, H3 * Dd);
  cvt(fwhh, whhf, H3 * Dd);
  cvt(rwih, wihr, H3 * Dd);
  cvt(rwhh, whhr, H3 * Dd);
  cvt(gate_w, gw16, Dd * 2 * Dd);
  cvt(out_w, ow16, Dd * Dd);

  ln_kernel<<<NROWS / 8, 256, 0, stream>>>(x, ln_g, ln_b, xn16);

  proj_kernel<<<dim3(NROWS / 16, 2), 512, 0, stream>>>(
      xn16, wihf, wihr, fbih, rbih, xp_f, xp_r);

  scan_kernel<<<4, 512, 0, stream>>>(xp_f, xp_r, whhf, whhr, fbhh, rbhh,
                                     h_f, h_r);

  gate_out_kernel<<<NROWS / 16, 512, 0, stream>>>(h_f, h_r, gw16, gate_b,
                                                  ow16, out_b, x, out);
}
